// MambaClassifier_62268435858020
// MI455X (gfx1250) — matrix operand order repaired and checked by reading, not compiled, not run on hardware
//
#include <hip/hip_runtime.h>
#include <hip/hip_bf16.h>
#include <math.h>

typedef __bf16 bf16_t;
typedef __attribute__((ext_vector_type(16))) __bf16 bf16x16;
typedef __attribute__((ext_vector_type(8)))  __bf16 bf16x8;
typedef __attribute__((ext_vector_type(8)))  float  f32x8;
typedef __attribute__((ext_vector_type(4)))  float  f32x4;

#define H_DIM   768
#define DI_DIM  500
#define NSTATE  16
#define RRANK   48
#define KCONV   4
#define NLYR    5
#define EPS_    1e-5f
#define BB      8
#define LLEN    1024
#define IN_DIM  4096
#define ML      (BB*LLEN)          // 8192 token rows
#define PROJ_N  (2*DI_DIM)         // 1000
#define SSM_N   (RRANK + 2*NSTATE) // 80

#define TM 128
#define TN 64
#define TK 32

__device__ __forceinline__ float sigmoidf_(float x) { return 1.0f / (1.0f + __expf(-x)); }
__device__ __forceinline__ float softplusf_(float x) { return (x > 20.0f) ? x : log1pf(__expf(x)); }

// ---------------------------------------------------------------------------
// GEMM: C[M,N] = act(A[M,K] @ B[K,N] + bias) + addsrc
// f32 in HBM -> bf16 at LDS-fill -> v_wmma_f32_16x16x32_bf16 -> f32 epilogue.
// 256 threads = 8 waves; 128x64 tile; K step 32; each wave owns 16 rows x 64 cols.
// Staging is branch-free (vector loads) whenever the tile is fully in range;
// bounds conditions are uniform so EXEC stays all-ones around the WMMAs.
// act: 0 = none, 1 = softplus
// ---------------------------------------------------------------------------
__global__ __launch_bounds__(256)
void gemm_bf16_wmma(const float* __restrict__ A, const float* __restrict__ Bw,
                    float* __restrict__ C, const float* __restrict__ bias,
                    const float* __restrict__ addsrc,
                    int M, int N, int K, int lda, int ldb, int ldc, int act)
{
    __shared__ __align__(16) bf16_t sA[TM][TK];   // [m][k]          8 KB
    __shared__ __align__(16) bf16_t sB[TN][TK];   // [n][k] (transp) 4 KB

    const int tid  = threadIdx.x;
    const int wave = tid >> 5;
    const int lane = tid & 31;
    const int row0 = blockIdx.x * TM;
    const int col0 = blockIdx.y * TN;

    const bool aRowsFull = (row0 + TM <= M);      // uniform
    const bool bColsFull = (col0 + TN <= N);      // uniform

    // Fast-path staging coordinates
    const int ar  = tid >> 1;                     // A: row 0..127
    const int ac0 = (tid & 1) * 16;               // A: k offset 0/16
    const int bc  = tid & 63;                     // B: n offset 0..63
    const int bkb = (tid >> 6) * 8;               // B: k offset 0,8,16,24

    f32x8 acc[4] = {f32x8{}, f32x8{}, f32x8{}, f32x8{}};

    for (int k0 = 0; k0 < K; k0 += TK) {
        const bool kFull = (k0 + TK <= K);        // uniform

        // ---- stage A tile (TM x TK) ----
        if (aRowsFull && kFull) {
            const float* src = A + (size_t)(row0 + ar) * lda + k0 + ac0;
            f32x4 f0 = *(const f32x4*)(src);
            f32x4 f1 = *(const f32x4*)(src + 4);
            f32x4 f2 = *(const f32x4*)(src + 8);
            f32x4 f3 = *(const f32x4*)(src + 12);
            bf16x8 h0, h1;
            #pragma unroll
            for (int j = 0; j < 4; ++j) { h0[j] = (bf16_t)f0[j]; h0[4 + j] = (bf16_t)f1[j]; }
            #pragma unroll
            for (int j = 0; j < 4; ++j) { h1[j] = (bf16_t)f2[j]; h1[4 + j] = (bf16_t)f3[j]; }
            *(bf16x8*)&sA[ar][ac0]     = h0;
            *(bf16x8*)&sA[ar][ac0 + 8] = h1;
            if (k0 + TK < K)
                __builtin_prefetch(src + TK, 0, 1);
        } else {
            #pragma unroll
            for (int i = 0; i < 16; ++i) {
                int e = i * 256 + tid;
                int r = e >> 5, c = e & 31;
                int gr = row0 + r, gk = k0 + c;
                float v = (gr < M && gk < K) ? A[(size_t)gr * lda + gk] : 0.0f;
                sA[r][c] = (bf16_t)v;
            }
        }

        // ---- stage B tile (TK x TN), transposed into [n][k] ----
        if (bColsFull && kFull) {
            const float* src = Bw + (size_t)(k0 + bkb) * ldb + col0 + bc;
            float v[8];
            #pragma unroll
            for (int j = 0; j < 8; ++j) v[j] = src[(size_t)j * ldb];
            bf16x8 h;
            #pragma unroll
            for (int j = 0; j < 8; ++j) h[j] = (bf16_t)v[j];
            *(bf16x8*)&sB[bc][bkb] = h;
            if (k0 + TK < K)
                __builtin_prefetch(src + (size_t)TK * ldb, 0, 1);
        } else {
            #pragma unroll
            for (int i = 0; i < 8; ++i) {
                int e = i * 256 + tid;
                int r = e >> 6, c = e & 63;        // r = k, c = n
                int gk = k0 + r, gn = col0 + c;
                float v = (gk < K && gn < N) ? Bw[(size_t)gk * ldb + gn] : 0.0f;
                sB[c][r] = (bf16_t)v;
            }
        }
        __syncthreads();

        // ---- fragments + WMMA ----
        // A frag (ISA 16-bit A 16x32 layout): m = lane&15, kbase = lane<16 ? 0 : 8,
        // e=0..7 -> k=kbase..kbase+7 ; e=8..15 -> k=16+kbase..
        {
            const int mr  = lane & 15;
            const int akb = (lane < 16) ? 0 : 8;
            union { bf16x16 v; bf16x8 h[2]; } ua;
            ua.h[0] = *(const bf16x8*)&sA[wave * 16 + mr][akb];
            ua.h[1] = *(const bf16x8*)&sA[wave * 16 + mr][akb + 16];

            // B frag: n = lane&15, kbase = lane<16 ? 0 : 8, same k order as A
            const int nc  = lane & 15;
            const int bkf = (lane < 16) ? 0 : 8;
            #pragma unroll
            for (int s = 0; s < 4; ++s) {
                union { bf16x16 v; bf16x8 h[2]; } ub;
                ub.h[0] = *(const bf16x8*)&sB[s * 16 + nc][bkf];
                ub.h[1] = *(const bf16x8*)&sB[s * 16 + nc][bkf + 16];
                acc[s] = __builtin_amdgcn_wmma_f32_16x16x32_bf16(
                    false, ua.v, false, ub.v, (short)0, acc[s], false, false);
            }
        }
        __syncthreads();
    }

    // ---- epilogue: C/D layout -> VGPR r holds M = r + 8*(lane>=16); N = lane&15
    const int nc = lane & 15;
    const int mbase = row0 + wave * 16 + ((lane >> 4) << 3);
    #pragma unroll
    for (int s = 0; s < 4; ++s) {
        int gc = col0 + s * 16 + nc;
        if (gc >= N) continue;
        float bv = bias ? bias[gc] : 0.0f;
        #pragma unroll
        for (int r = 0; r < 8; ++r) {
            int gr = mbase + r;
            if (gr >= M) continue;
            float v = acc[s][r] + bv;
            if (act == 1) v = softplusf_(v);
            if (addsrc) v += addsrc[(size_t)gr * ldc + gc];
            C[(size_t)gr * ldc + gc] = v;
        }
    }
}

// ---------------------------------------------------------------------------
// RMSNorm over H=768 per token row
// ---------------------------------------------------------------------------
__global__ __launch_bounds__(256)
void rmsnorm_kernel(const float* __restrict__ x, const float* __restrict__ w,
                    float* __restrict__ out)
{
    __shared__ float red[256];
    const size_t row = blockIdx.x;
    const float* xr = x + row * H_DIM;
    float ss = 0.0f;
    for (int c = threadIdx.x; c < H_DIM; c += 256) { float v = xr[c]; ss += v * v; }
    red[threadIdx.x] = ss;
    __syncthreads();
    for (int s = 128; s > 0; s >>= 1) {
        if (threadIdx.x < s) red[threadIdx.x] += red[threadIdx.x + s];
        __syncthreads();
    }
    float scale = rsqrtf(red[0] / (float)H_DIM + EPS_);
    for (int c = threadIdx.x; c < H_DIM; c += 256)
        out[row * H_DIM + c] = xr[c] * scale * w[c];
}

// ---------------------------------------------------------------------------
// Depthwise causal conv (K=4) over u = proj[:, :500], + bias, + SiLU
// ---------------------------------------------------------------------------
__global__ __launch_bounds__(256)
void conv_silu_kernel(const float* __restrict__ proj, const float* __restrict__ cw,
                      const float* __restrict__ cb, float* __restrict__ uact)
{
    int idx = blockIdx.x * blockDim.x + threadIdx.x;
    if (idx >= ML * DI_DIM) return;
    int d  = idx % DI_DIM;
    int bl = idx / DI_DIM;
    int l  = bl % LLEN;
    float acc = cb[d];
    #pragma unroll
    for (int k = 0; k < KCONV; ++k) {
        int ls = l + k - (KCONV - 1);
        if (ls >= 0) acc += proj[(size_t)(bl + k - (KCONV - 1)) * PROJ_N + d] * cw[d * KCONV + k];
    }
    uact[idx] = acc * sigmoidf_(acc);
}

// ---------------------------------------------------------------------------
// Selective scan: one thread per (b, d); N=16 state in registers.
// B/C vectors for 64 timesteps are staged into LDS with the gfx1250 async
// global->LDS DMA path (GLOBAL_LOAD_ASYNC_TO_LDS_B128, ASYNCcnt): raw f32
// bytes, no conversion, never touches VGPRs. Each of the 512 threads moves one
// 16-byte quad (8 KB total per chunk), drains with s_wait_asynccnt 0, then
// barriers. Fuses +u*D and *silu(gate).
// ---------------------------------------------------------------------------
__global__ __launch_bounds__(512)
void mamba_scan(const float* __restrict__ u, const float* __restrict__ dt,
                const float* __restrict__ ssm, const float* __restrict__ proj,
                const float* __restrict__ A_log, const float* __restrict__ Dp,
                float* __restrict__ y)
{
    __shared__ __align__(16) float bcs[64][32];   // [t][0..15]=B, [t][16..31]=C
    const int b = blockIdx.x;
    const int d = threadIdx.x;
    const bool live = (d < DI_DIM);

    // async-staging coordinates: one b128 per thread per chunk
    const int sr  = threadIdx.x >> 3;             // t row 0..63
    const int sc0 = (threadIdx.x & 7) * 4;        // float col 0,4,..,28

    float Ac[NSTATE], h[NSTATE];
    float dcoef = live ? Dp[d] : 0.0f;
    #pragma unroll
    for (int n = 0; n < NSTATE; ++n) {
        Ac[n] = live ? -__expf(A_log[d * NSTATE + n]) : 0.0f;
        h[n]  = 0.0f;
    }

    for (int lc = 0; lc < LLEN; lc += 64) {
        {
            const float* gsrc = ssm + (size_t)(b * LLEN + lc + sr) * SSM_N + RRANK + sc0;
            unsigned ldsdst = (unsigned)(uintptr_t)&bcs[sr][sc0];  // low 32 bits = LDS offset
            asm volatile("global_load_async_to_lds_b128 %0, %1, off"
                         :: "v"(ldsdst), "v"(gsrc) : "memory");
        }
        asm volatile("s_wait_asynccnt 0" ::: "memory");
        __syncthreads();
        if (live) {
            for (int t = 0; t < 64; ++t) {
                size_t idx = (size_t)(b * LLEN + lc + t) * DI_DIM + d;
                float dtv = dt[idx];
                float uv  = u[idx];
                float sum = 0.0f;
                #pragma unroll
                for (int n = 0; n < NSTATE; ++n) {
                    h[n] = __expf(dtv * Ac[n]) * h[n] + dtv * bcs[t][n] * uv;
                    sum += h[n] * bcs[t][16 + n];
                }
                float g = proj[(size_t)(b * LLEN + lc + t) * PROJ_N + DI_DIM + d];
                y[idx] = (sum + uv * dcoef) * (g * sigmoidf_(g));
            }
        }
        __syncthreads();
    }
}

// ---------------------------------------------------------------------------
// Final: rmsnorm(last token) . W_cls  -> logits[b]
// ---------------------------------------------------------------------------
__global__ __launch_bounds__(256)
void final_kernel(const float* __restrict__ x, const float* __restrict__ wn,
                  const float* __restrict__ wcls, float* __restrict__ out)
{
    __shared__ float red[256];
    const int b = blockIdx.x;
    const float* xr = x + ((size_t)b * LLEN + (LLEN - 1)) * H_DIM;
    float ss = 0.0f;
    for (int c = threadIdx.x; c < H_DIM; c += 256) { float v = xr[c]; ss += v * v; }
    red[threadIdx.x] = ss;
    __syncthreads();
    for (int s = 128; s > 0; s >>= 1) {
        if (threadIdx.x < s) red[threadIdx.x] += red[threadIdx.x + s];
        __syncthreads();
    }
    float scale = rsqrtf(red[0] / (float)H_DIM + EPS_);
    __syncthreads();
    float dot = 0.0f;
    for (int c = threadIdx.x; c < H_DIM; c += 256)
        dot += xr[c] * scale * wn[c] * wcls[c];
    red[threadIdx.x] = dot;
    __syncthreads();
    for (int s = 128; s > 0; s >>= 1) {
        if (threadIdx.x < s) red[threadIdx.x] += red[threadIdx.x + s];
        __syncthreads();
    }
    if (threadIdx.x == 0) out[b] = red[0];
}

// ---------------------------------------------------------------------------
extern "C" void kernel_launch(void* const* d_in, const int* in_sizes, int n_in,
                              void* d_out, int out_size, void* d_ws, size_t ws_size,
                              hipStream_t stream)
{
    (void)in_sizes; (void)n_in; (void)out_size; (void)ws_size;
    const float* inputs   = (const float*)d_in[0];
    const float* W_down   = (const float*)d_in[1];
    const float* in_proj  = (const float*)d_in[2];
    const float* conv_w   = (const float*)d_in[3];
    const float* conv_b   = (const float*)d_in[4];
    const float* x_proj   = (const float*)d_in[5];
    const float* dt_w     = (const float*)d_in[6];
    const float* dt_b     = (const float*)d_in[7];
    const float* A_log    = (const float*)d_in[8];
    const float* D_param  = (const float*)d_in[9];
    const float* out_w    = (const float*)d_in[10];
    const float* norm_w   = (const float*)d_in[11];
    const float* norm_f_w = (const float*)d_in[12];
    const float* W_cls    = (const float*)d_in[13];

    float* ws   = (float*)d_ws;
    float* x    = ws;                                   // ML*768
    float* xn   = x    + (size_t)ML * H_DIM;            // ML*768
    float* proj = xn   + (size_t)ML * H_DIM;            // ML*1000
    float* uact = proj + (size_t)ML * PROJ_N;           // ML*500
    float* ssm  = uact + (size_t)ML * DI_DIM;           // ML*80
    float* dtb  = ssm  + (size_t)ML * SSM_N;            // ML*500
    float* yb   = dtb  + (size_t)ML * DI_DIM;           // ML*500

    auto gemm = [&](const float* A, const float* Bw, float* C, const float* bias,
                    const float* add, int M, int N, int K, int lda, int ldb, int ldc, int act) {
        dim3 grid((M + TM - 1) / TM, (N + TN - 1) / TN);
        hipLaunchKernelGGL(gemm_bf16_wmma, grid, dim3(256), 0, stream,
                           A, Bw, C, bias, add, M, N, K, lda, ldb, ldc, act);
    };

    // x = inputs @ W_down
    gemm(inputs, W_down, x, nullptr, nullptr, ML, H_DIM, IN_DIM, IN_DIM, H_DIM, H_DIM, 0);

    for (int i = 0; i < NLYR; ++i) {
        hipLaunchKernelGGL(rmsnorm_kernel, dim3(ML), dim3(256), 0, stream,
                           x, norm_w + (size_t)i * H_DIM, xn);
        // proj = xn @ in_proj[i]
        gemm(xn, in_proj + (size_t)i * H_DIM * PROJ_N, proj, nullptr, nullptr,
             ML, PROJ_N, H_DIM, H_DIM, PROJ_N, PROJ_N, 0);
        // u = silu(conv(u) + b)
        {
            int total = ML * DI_DIM;
            hipLaunchKernelGGL(conv_silu_kernel, dim3((total + 255) / 256), dim3(256), 0, stream,
                               proj, conv_w + (size_t)i * DI_DIM * KCONV,
                               conv_b + (size_t)i * DI_DIM, uact);
        }
        // ssm = u @ x_proj[i]
        gemm(uact, x_proj + (size_t)i * DI_DIM * SSM_N, ssm, nullptr, nullptr,
             ML, SSM_N, DI_DIM, DI_DIM, SSM_N, SSM_N, 0);
        // dt = softplus(ssm[:, :48] @ dt_w[i] + dt_b[i])
        gemm(ssm, dt_w + (size_t)i * RRANK * DI_DIM, dtb, dt_b + (size_t)i * DI_DIM, nullptr,
             ML, DI_DIM, RRANK, SSM_N, DI_DIM, DI_DIM, 1);
        // selective scan (fuses +u*D and *silu(gate))
        hipLaunchKernelGGL(mamba_scan, dim3(BB), dim3(512), 0, stream,
                           uact, dtb, ssm, proj,
                           A_log + (size_t)i * DI_DIM * NSTATE,
                           D_param + (size_t)i * DI_DIM, yb);
        // x = x + y @ out_w[i]   (residual fused into GEMM epilogue)
        gemm(yb, out_w + (size_t)i * DI_DIM * H_DIM, x, nullptr, x,
             ML, H_DIM, DI_DIM, DI_DIM, H_DIM, H_DIM, 0);
    }

    hipLaunchKernelGGL(final_kernel, dim3(BB), dim3(256), 0, stream,
                       x, norm_f_w, W_cls, (float*)d_out);
}
